// VocabGraphConvolution_25675314496035
// MI455X (gfx1250) — matrix operand order repaired and checked by reading, not compiled, not run on hardware
//
#include <hip/hip_runtime.h>
#include <hip/hip_bf16.h>

// ---------------------------------------------------------------------------
// VocabGraphConvolution on MI455X (gfx1250)
//
//   H_sum[v,h]   = sum_a sum_nnz vals[a,e] * W[a, cols[a,e], h]
//   fused[d,h]   = sum_v X[d,v] * H_sum[v,h]     <- 15.7 GFLOP, 245.8 MB of X
//   out[d,o]     = sum_h fused[d,h]*fc_w[o,h] + fc_b[o]
//
// Roofline: stage 2 is HBM-bound (245.8MB / 23.3TB/s ~ 10.5us) provided the
// math runs on the matrix pipes: v_wmma_f32_16x16x32_bf16 (f32 accumulate).
// X is read ONCE as f32 with non-temporal hint (streamed, never re-read, must
// not evict the L2-resident 15.4MB H_sum), converted to bf16 in-register, and
// staged in LDS pre-swizzled into the WMMA fragment layout so the inner loop
// is pure ds_load_b128 + v_wmma. The 16-wide K remainder (30000 % 32) is
// peeled out of the hot loop so no guards/branches remain in it.
// ---------------------------------------------------------------------------

#define V_DIM   30000
#define NNZ_E   480000
#define NADJ    2
#define HID     128
#define OUT_DIM 128
#define D_DIM   2048
#define KSPLIT  8
#define KFULL   (V_DIM / 32)          // 937 full 32-wide K tiles
#define KTAIL   (V_DIM - KFULL * 32)  // 16 remaining K values

typedef __attribute__((ext_vector_type(16))) __bf16 v16bf;
typedef __attribute__((ext_vector_type(8)))  float  v8f;
typedef __attribute__((ext_vector_type(4)))  float  f4;

__device__ __forceinline__ unsigned short f32_bf16(float f) {
  unsigned u = __builtin_bit_cast(unsigned, f);
  u += 0x7FFFu + ((u >> 16) & 1u);        // round-to-nearest-even
  return (unsigned short)(u >> 16);
}

// pack 8 f32 -> 8 bf16 and store as one 16B LDS write
__device__ __forceinline__ void store8bf16(unsigned short* dst, const float* f) {
  union { uint4 q; unsigned short h[8]; } pk;
  #pragma unroll
  for (int i = 0; i < 8; ++i) pk.h[i] = f32_bf16(f[i]);
  *reinterpret_cast<uint4*>(dst) = pk.q;
}

// ---------------------------------------------------------------- zero scratch
__global__ void zero_ws(float* __restrict__ p, int n) {
  int i = blockIdx.x * blockDim.x + threadIdx.x;
  int stride = gridDim.x * blockDim.x;
  for (; i < n; i += stride) p[i] = 0.0f;
}

// ------------------------------------------------- COO scatter: H_sum = sum A_i W_i
// One thread per (edge, 4-wide hid chunk); consecutive lanes read consecutive
// quads of one W row (coalesced 512B); f32 global atomics into L2-resident H.
__global__ void spmm_scatter(const int* __restrict__ rows,
                             const int* __restrict__ cols,
                             const float* __restrict__ vals,
                             const float* __restrict__ W,
                             float* __restrict__ Hsum) {
  long long tid = (long long)blockIdx.x * blockDim.x + threadIdx.x;
  const long long total = (long long)NADJ * NNZ_E * (HID / 4);
  if (tid >= total) return;
  int hq = (int)(tid & 31);            // HID/4 == 32
  long long e = tid >> 5;              // global edge index
  int a = (int)(e / NNZ_E);
  int i = (int)(e % NNZ_E);
  int r = rows[a * NNZ_E + i];
  int c = cols[a * NNZ_E + i];
  float v = vals[a * NNZ_E + i];
  const float4 w = *reinterpret_cast<const float4*>(
      W + ((long long)a * V_DIM + c) * HID + hq * 4);
  float* dst = Hsum + (long long)r * HID + hq * 4;
  atomicAdd(dst + 0, v * w.x);
  atomicAdd(dst + 1, v * w.y);
  atomicAdd(dst + 2, v * w.z);
  atomicAdd(dst + 3, v * w.w);
}

// ---------------------------------------------------------------- main GEMM
// fused[2048,128] += X[2048,30000] @ Hsum[30000,128]   (K split over grid.y)
//
// Block: 256 threads (8 waves), 64(M) x 128(N) output tile; full N per block
// so X is read exactly once from HBM. Fragment slot maps used here (contiguous k, not the ISA table order):
//   A(16x32): lane = m + 16*(k>>4);      half = (k&15)
//   B(32x16): lane = n + 16*(k>>4);      half = (k&15)
__global__ void __launch_bounds__(256)
gemm_xh(const float* __restrict__ X, const float* __restrict__ H,
        float* __restrict__ fused) {
  // A fragments: 4 strips * 512 halves ; B fragments: 8 ntiles * 512 halves
  __shared__ __align__(16) unsigned short smem[6144];   // 12 KB

  const int tid  = threadIdx.x;
  const int lane = tid & 31;
  const int wave = tid >> 5;
  const int dBase = blockIdx.x * 64;

  const int tilesPer = (KFULL + KSPLIT - 1) / KSPLIT;           // 118
  const int t0   = blockIdx.y * tilesPer;
  const int tFul = (t0 + tilesPer < KFULL) ? (t0 + tilesPer) : KFULL;
  const bool hasTail = (blockIdx.y == KSPLIT - 1);

  const int strip = wave >> 1;          // M strip 0..3
  const int nhalf = wave & 1;           // N half 0..1

  const v8f vzero = {0.f, 0.f, 0.f, 0.f, 0.f, 0.f, 0.f, 0.f};
  v8f acc[4] = {vzero, vzero, vzero, vzero};

  // A-loader mapping: thread -> (row 0..63, k chunk of 8)
  const int arow = tid >> 2;
  const int ak0  = (tid & 3) * 8;
  const int a_m      = arow & 15;
  const int a_strp   = arow >> 4;
  const int a_lane   = a_m + 16 * (ak0 >> 4);
  const int a_hbase  = ak0 & 15;
  unsigned short* aDst = &smem[a_strp * 512 + a_lane * 16 + a_hbase];
  const float* aSrcRow = X + (long long)(dBase + arow) * V_DIM + ak0;

  // B-loader mapping: thread -> (n 0..127, k half 0..1)
  const int bn  = tid & 127;
  const int bkh = tid >> 7;
  const int b_lane = (bn & 15) + 16 * bkh;
  unsigned short* bDst = &smem[2048 + (bn >> 4) * 512 + b_lane * 16];
  const float* bSrcCol = H + (long long)(bkh * 16) * HID + bn;

  // --------------------- hot loop: full 32-wide K tiles, zero branches ----
  for (int t = t0; t < tFul; ++t) {
    const int kBase = t * 32;
    {   // stage A (64x32): streamed non-temporal, X never re-read
      const f4* src = reinterpret_cast<const f4*>(aSrcRow + kBase);
      f4 a0 = __builtin_nontemporal_load(src);
      f4 a1 = __builtin_nontemporal_load(src + 1);
      float f[8] = {a0[0], a0[1], a0[2], a0[3], a1[0], a1[1], a1[2], a1[3]};
      store8bf16(aDst, f);
    }
    {   // stage B (32x128): regular loads, keep H hot in L2
      const float* src = bSrcCol + (long long)kBase * HID;
      float f0[8], f1[8];
      #pragma unroll
      for (int i = 0; i < 8; ++i) f0[i] = src[i * HID];
      #pragma unroll
      for (int i = 0; i < 8; ++i) f1[i] = src[(8 + i) * HID];
      store8bf16(bDst,     f0);
      store8bf16(bDst + 8, f1);
    }
    __syncthreads();
    {   // matrix cores: 4 x v_wmma_f32_16x16x32_bf16 per wave
      union { uint4 q[2]; v16bf v; } A;
      const uint4* ap =
          reinterpret_cast<const uint4*>(&smem[strip * 512 + lane * 16]);
      A.q[0] = ap[0];
      A.q[1] = ap[1];
      #pragma unroll
      for (int i = 0; i < 4; ++i) {
        const int nt = nhalf * 4 + i;
        union { uint4 q[2]; v16bf v; } B;
        const uint4* bp =
            reinterpret_cast<const uint4*>(&smem[2048 + nt * 512 + lane * 16]);
        B.q[0] = bp[0];
        B.q[1] = bp[1];
        acc[i] = __builtin_amdgcn_wmma_f32_16x16x32_bf16(
            false, A.v, false, B.v, (short)0, acc[i], false, false);
      }
    }
    __syncthreads();
  }

  // --------------------- peeled 16-wide K tail (last split only) ----------
  // Valid k are kBase..kBase+15: A chunks ak0>=16 and B half bkh==1 are
  // statically zero -> no per-element guards, no OOB reads.
  if (hasTail) {
    const int kBase = KFULL * 32;                 // 29984
    {
      float f[8] = {0, 0, 0, 0, 0, 0, 0, 0};
      if (ak0 < KTAIL) {
        const f4* src = reinterpret_cast<const f4*>(aSrcRow + kBase);
        f4 a0 = __builtin_nontemporal_load(src);
        f4 a1 = __builtin_nontemporal_load(src + 1);
        f[0] = a0[0]; f[1] = a0[1]; f[2] = a0[2]; f[3] = a0[3];
        f[4] = a1[0]; f[5] = a1[1]; f[6] = a1[2]; f[7] = a1[3];
      }
      store8bf16(aDst, f);
    }
    {
      float f0[8] = {0, 0, 0, 0, 0, 0, 0, 0};
      float f1[8] = {0, 0, 0, 0, 0, 0, 0, 0};
      if (bkh == 0) {
        const float* src = bSrcCol + (long long)kBase * HID;
        #pragma unroll
        for (int i = 0; i < 8; ++i) f0[i] = src[i * HID];
        #pragma unroll
        for (int i = 0; i < 8; ++i) f1[i] = src[(8 + i) * HID];
      }
      store8bf16(bDst,     f0);
      store8bf16(bDst + 8, f1);
    }
    __syncthreads();
    {
      union { uint4 q[2]; v16bf v; } A;
      const uint4* ap =
          reinterpret_cast<const uint4*>(&smem[strip * 512 + lane * 16]);
      A.q[0] = ap[0];
      A.q[1] = ap[1];
      #pragma unroll
      for (int i = 0; i < 4; ++i) {
        const int nt = nhalf * 4 + i;
        union { uint4 q[2]; v16bf v; } B;
        const uint4* bp =
            reinterpret_cast<const uint4*>(&smem[2048 + nt * 512 + lane * 16]);
        B.q[0] = bp[0];
        B.q[1] = bp[1];
        acc[i] = __builtin_amdgcn_wmma_f32_16x16x32_bf16(
            false, A.v, false, B.v, (short)0, acc[i], false, false);
      }
    }
  }

  // ---- K-split reduction: atomic f32 accumulate into fused[2048,128] ----
  // C/D layout: lanes 0-15 -> M = vgpr, N = lane; lanes 16-31 -> M = vgpr+8.
  const int mrow  = dBase + strip * 16 + (lane >> 4) * 8;
  const int ncol0 = nhalf * 64 + (lane & 15);
  #pragma unroll
  for (int i = 0; i < 4; ++i) {
    const int n = ncol0 + i * 16;
    #pragma unroll
    for (int r = 0; r < 8; ++r)
      atomicAdd(&fused[(mrow + r) * HID + n], acc[i][r]);
  }
}

// ---------------------------------------------------------------- final FC
__global__ void fc_kernel(const float* __restrict__ fused,
                          const float* __restrict__ Wfc,
                          const float* __restrict__ bfc,
                          float* __restrict__ out) {
  __shared__ float s[HID];
  const int d = blockIdx.x;
  const int o = threadIdx.x;
  s[o] = fused[d * HID + o];
  __syncthreads();
  float acc = bfc[o];
  #pragma unroll 8
  for (int k = 0; k < HID; ++k) acc += s[k] * Wfc[o * HID + k];
  out[d * OUT_DIM + o] = acc;
}

// ---------------------------------------------------------------- launcher
extern "C" void kernel_launch(void* const* d_in, const int* in_sizes, int n_in,
                              void* d_out, int out_size, void* d_ws,
                              size_t ws_size, hipStream_t stream) {
  const int*   adj_rows = (const int*)d_in[0];
  const int*   adj_cols = (const int*)d_in[1];
  const float* adj_vals = (const float*)d_in[2];
  const float* X        = (const float*)d_in[3];
  const float* W        = (const float*)d_in[4];
  const float* fcw      = (const float*)d_in[5];
  const float* fcb      = (const float*)d_in[6];
  float*       out      = (float*)d_out;

  float* Hsum  = (float*)d_ws;                      // V*HID f32 (15.36 MB)
  float* fused = Hsum + (size_t)V_DIM * HID;        // D*HID f32 (1 MB)

  // zero scratch (ws is poisoned; must re-zero every call for determinism)
  const int zeroN = V_DIM * HID + D_DIM * HID;
  hipLaunchKernelGGL(zero_ws, dim3(2048), dim3(256), 0, stream, Hsum, zeroN);

  // sparse scatter of both adjacencies into one H_sum
  const long long spmmThreads = (long long)NADJ * NNZ_E * (HID / 4);
  const int spmmBlocks = (int)((spmmThreads + 255) / 256);
  hipLaunchKernelGGL(spmm_scatter, dim3(spmmBlocks), dim3(256), 0, stream,
                     adj_rows, adj_cols, adj_vals, W, Hsum);

  // bandwidth-bound WMMA GEMM, K split 8 ways -> 256 workgroups
  hipLaunchKernelGGL(gemm_xh, dim3(D_DIM / 64, KSPLIT), dim3(256), 0, stream,
                     X, Hsum, fused);

  // tiny FC epilogue
  hipLaunchKernelGGL(fc_kernel, dim3(D_DIM), dim3(OUT_DIM), 0, stream,
                     fused, fcw, fcb, out);
}
